// KaplanAttention_438086664882
// MI455X (gfx1250) — compile-verified, matrix operand order checked by reading, not run on hardware
//
#include <hip/hip_runtime.h>
#include <hip/hip_bf16.h>

// ---------------------------------------------------------------------------
// Causal MHA forward for MI455X (gfx1250, wave32, WMMA).
// fp32->bf16 | QKV proj (bf16 WMMA, double-buffered K loop) | flash attention
// (32 queries/wave, WMMA-computed softmax denominator, mask-peeled) | out proj.
// ---------------------------------------------------------------------------

typedef __bf16 bf16_t;
typedef __attribute__((ext_vector_type(16))) __bf16 v16bf;
typedef __attribute__((ext_vector_type(8)))  __bf16 v8bf;
typedef __attribute__((ext_vector_type(8)))  float  v8f;
typedef __attribute__((ext_vector_type(4)))  float  v4f;

#define D_MODEL 1024
#define N_HEADS 16
#define D_K     64
#define SEQ     2048
#define BATCH   2
#define M_TOTAL (BATCH * SEQ)   // 4096 rows

// softmax scale folded into Q at projection time: 1/sqrt(64) * log2(e)
#define SM_SCALE (0.125f * 1.44269504088896340736f)

// ---- fp32 -> bf16: native convert (RNE fptrunc, hardware cvt on gfx1250) --
__device__ __forceinline__ bf16_t f2bf(float f) { return (bf16_t)f; }

// single v_max_num_f32: no canonicalize (fmaxf) and no cmp+cndmask (ternary)
__device__ __forceinline__ float fast_max(float a, float b) {
    float d;
    asm("v_max_num_f32_e32 %0, %1, %2" : "=v"(d) : "v"(a), "v"(b));
    return d;
}

__device__ __forceinline__ v16bf ones_frag() {
    union { unsigned u[8]; v16bf v; } o;
#pragma unroll
    for (int i = 0; i < 8; ++i) o.u[i] = 0x3F803F80u;   // two bf16 1.0s
    return o.v;
}

// ---- 16x32 bf16 A-fragment load (row-major source, K-contiguous) ----------
// Lanes 0-15 hold row M=lane with K blocks [0..8)/[16..24); lanes 16-31 hold
// the same rows with K [8..16)/[24..32). B fragments for C = A*B^T are just
// A-fragments of the row-major B, so all operands are 2x128-bit row loads.
__device__ __forceinline__ v16bf load_frag(const bf16_t* base, int ld, int lane) {
    const int r = lane & 15, half = lane >> 4;
    const bf16_t* p = base + (size_t)r * ld + half * 8;
    v8bf lo = *reinterpret_cast<const v8bf*>(p);
    v8bf hi = *reinterpret_cast<const v8bf*>(p + 16);
    return __builtin_shufflevector(lo, hi, 0,1,2,3,4,5,6,7,8,9,10,11,12,13,14,15);
}

__device__ __forceinline__ v8f wmma_bf16(v16bf a, v16bf b, v8f c) {
    return __builtin_amdgcn_wmma_f32_16x16x32_bf16(
        false, a, false, b, (short)0, c, false, false);
}

// ---------------------------------------------------------------------------
// Kernel 0: fp32 -> bf16 convert, 8 elements/thread (2x b128 in, 1x b128 out)
// ---------------------------------------------------------------------------
__global__ void cvt_f32_bf16(const float* __restrict__ src, bf16_t* __restrict__ dst, int n8) {
    int i = blockIdx.x * blockDim.x + threadIdx.x;
    int stride = gridDim.x * blockDim.x;
    for (; i < n8; i += stride) {
        const v4f* s = reinterpret_cast<const v4f*>(src + (size_t)i * 8);
        v4f x0 = s[0], x1 = s[1];
        v8bf o;
#pragma unroll
        for (int e = 0; e < 4; ++e) { o[e] = f2bf(x0[e]); o[4 + e] = f2bf(x1[e]); }
        *reinterpret_cast<v8bf*>(dst + (size_t)i * 8) = o;
    }
}

// ---------------------------------------------------------------------------
// GEMM core: 32(M) x 64(N) per wave, K double-buffered so WMMAs on one
// register set overlap the loads of the other (partial loadcnt waits).
// ---------------------------------------------------------------------------
__device__ __forceinline__ void gemm_tile_32x64(
        const bf16_t* __restrict__ Arow0, const bf16_t* __restrict__ Arow1,
        const bf16_t* __restrict__ Wrow, int lane, v8f acc[2][4]) {
    v16bf a0[2], b0[4], a1[2], b1[4];
    a0[0] = load_frag(Arow0, D_MODEL, lane);
    a0[1] = load_frag(Arow1, D_MODEL, lane);
#pragma unroll
    for (int c = 0; c < 4; ++c)
        b0[c] = load_frag(Wrow + (size_t)c * 16 * D_MODEL, D_MODEL, lane);

    for (int kk = 0; kk < D_MODEL; kk += 64) {
        __builtin_prefetch(Arow0 + kk + 128, 0, 1);     // -> global_prefetch_b8
        __builtin_prefetch(Wrow  + kk + 128, 0, 1);
        a1[0] = load_frag(Arow0 + kk + 32, D_MODEL, lane);
        a1[1] = load_frag(Arow1 + kk + 32, D_MODEL, lane);
#pragma unroll
        for (int c = 0; c < 4; ++c)
            b1[c] = load_frag(Wrow + (size_t)c * 16 * D_MODEL + kk + 32, D_MODEL, lane);
#pragma unroll
        for (int mi = 0; mi < 2; ++mi)
#pragma unroll
            for (int c = 0; c < 4; ++c)
                acc[mi][c] = wmma_bf16(mi ? a0[1] : a0[0], b0[c], acc[mi][c]);
        if (kk + 64 < D_MODEL) {
            a0[0] = load_frag(Arow0 + kk + 64, D_MODEL, lane);
            a0[1] = load_frag(Arow1 + kk + 64, D_MODEL, lane);
#pragma unroll
            for (int c = 0; c < 4; ++c)
                b0[c] = load_frag(Wrow + (size_t)c * 16 * D_MODEL + kk + 64, D_MODEL, lane);
        }
#pragma unroll
        for (int mi = 0; mi < 2; ++mi)
#pragma unroll
            for (int c = 0; c < 4; ++c)
                acc[mi][c] = wmma_bf16(mi ? a1[1] : a1[0], b1[c], acc[mi][c]);
    }
}

// ---------------------------------------------------------------------------
// Kernel 1: projection GEMM  C = scale * (X @ W^T), bf16 out.
// grid = (N_HEADS, M_TOTAL/128); block = 128 (4 waves x 32 rows).
// VT=false: out[bh][s][d] (Q,K).  VT=true: out[bh][d][s] (V transposed);
// the transposed layout makes a lane's 8 row-values s-contiguous -> b128 store.
// ---------------------------------------------------------------------------
template<bool VT>
__global__ __launch_bounds__(128)
void proj_gemm(const bf16_t* __restrict__ A, const bf16_t* __restrict__ W,
               bf16_t* __restrict__ out, float out_scale) {
    const int lane = threadIdx.x & 31;
    const int wave = threadIdx.x >> 5;
    const int m0 = blockIdx.y * 128 + wave * 32;
    const int h  = blockIdx.x;
    const int n0 = h * D_K;

    v8f acc[2][4] = {};
    gemm_tile_32x64(A + (size_t)m0 * D_MODEL, A + (size_t)(m0 + 16) * D_MODEL,
                    W + (size_t)n0 * D_MODEL, lane, acc);

    const int bidx = m0 >> 11;
    const int s    = m0 & (SEQ - 1);
    const int bh   = bidx * N_HEADS + h;
    const int n = lane & 15, half = lane >> 4;
#pragma unroll
    for (int mi = 0; mi < 2; ++mi)
#pragma unroll
        for (int c = 0; c < 4; ++c) {
            if (VT) {
                // row d = c*16+n, cols s+mi*16+8*half+j contiguous over j
                const int d = c * 16 + n;
                v8bf pv;
#pragma unroll
                for (int j = 0; j < 8; ++j) pv[j] = f2bf(acc[mi][c][j] * out_scale);
                *reinterpret_cast<v8bf*>(out + ((size_t)bh * D_K + d) * SEQ
                                             + s + mi * 16 + 8 * half) = pv;
            } else {
#pragma unroll
                for (int j = 0; j < 8; ++j) {
                    const int m = mi * 16 + j + 8 * half;
                    out[((size_t)bh * SEQ + (s + m)) * D_K + c * 16 + n] =
                        f2bf(acc[mi][c][j] * out_scale);
                }
            }
        }
}

// ---------------------------------------------------------------------------
// Flash-attention key-block body (32 keys x 32 queries). MASKED only for the
// final diagonal block. The softmax denominator is accumulated with a WMMA
// against a ones B-fragment (accS = running P @ 1, rescaled like acc), so the
// only cross-lane reduction left is the row-max tree.
// ---------------------------------------------------------------------------
template<bool MASKED>
__device__ __forceinline__ void attn_block(
        int j0, int q0, int lane,
        const bf16_t* __restrict__ Kh, const bf16_t* __restrict__ Vh,
        const v16bf aq[2][2], v16bf bones, bf16_t* ldsP,
        float rmax[2][8], v8f acc[2][4], v8f accS[2]) {
    const int half = lane >> 4, n = lane & 15;

    v16bf bk[4], bv[4];
    bk[0] = load_frag(Kh + (size_t)(j0     ) * D_K +  0, D_K, lane);
    bk[1] = load_frag(Kh + (size_t)(j0     ) * D_K + 32, D_K, lane);
    bk[2] = load_frag(Kh + (size_t)(j0 + 16) * D_K +  0, D_K, lane);
    bk[3] = load_frag(Kh + (size_t)(j0 + 16) * D_K + 32, D_K, lane);

#pragma unroll
    for (int mi = 0; mi < 2; ++mi) {
        v8f c0 = {}, c1 = {};
        c0 = wmma_bf16(aq[mi][0], bk[0], c0);
        c0 = wmma_bf16(aq[mi][1], bk[1], c0);
        c1 = wmma_bf16(aq[mi][0], bk[2], c1);
        c1 = wmma_bf16(aq[mi][1], bk[3], c1);

        // Issue V fragment loads after the first score WMMAs: latency hides
        // under both softmax phases without spanning the whole block body
        // (keeps VGPR live ranges short enough to stay under the MSB limit).
        if (mi == 0) {
#pragma unroll
            for (int c = 0; c < 4; ++c)
                bv[c] = load_frag(Vh + (size_t)(c * 16) * SEQ + j0, SEQ, lane);
        }

        v8f alphas;
#pragma unroll
        for (int j = 0; j < 8; ++j) {
            float s0 = c0[j];                 // already in scaled log2 domain
            float s1 = c1[j];
            if (MASKED) {
                const int qrow = q0 + mi * 16 + j + 8 * half;
                if (j0 + n      > qrow) s0 = -__builtin_inff();
                if (j0 + 16 + n > qrow) s1 = -__builtin_inff();
            }
            float mb = fast_max(s0, s1);
#pragma unroll
            for (int off = 1; off < 16; off <<= 1)
                mb = fast_max(mb, __shfl_xor(mb, off, 32));
            const float mnew = fast_max(rmax[mi][j], mb);
            alphas[j] = exp2f(rmax[mi][j] - mnew);
            rmax[mi][j] = mnew;
            const int m = mi * 16 + j + 8 * half;
            ldsP[m * 32 + n]      = f2bf(exp2f(s0 - mnew));
            ldsP[m * 32 + 16 + n] = f2bf(exp2f(s1 - mnew));
        }
#pragma unroll
        for (int c = 0; c < 4; ++c) acc[mi][c] *= alphas;   // v8f * v8f
        accS[mi] *= alphas;
    }

    // Per-wave DS ordering; drain P stores before the A-fragment reload.
    asm volatile("s_wait_dscnt 0" ::: "memory");

#pragma unroll
    for (int mi = 0; mi < 2; ++mi) {
        const v16bf ap = load_frag(ldsP + mi * 16 * 32, 32, lane);
#pragma unroll
        for (int c = 0; c < 4; ++c)
            acc[mi][c] = wmma_bf16(ap, bv[c], acc[mi][c]);
        accS[mi] = wmma_bf16(ap, bones, accS[mi]);          // row sums via WMMA
    }
    asm volatile("s_wait_dscnt 0" ::: "memory");  // reads done before next P store
}

// ---------------------------------------------------------------------------
// Kernel 2: causal flash attention. grid = (SEQ/32/4, BATCH*N_HEADS);
// block = 128 = 4 independent waves (32 queries each, private LDS P tile,
// no cross-wave barriers so divergent causal trip counts are safe).
// ---------------------------------------------------------------------------
__global__ __launch_bounds__(128)
void flash_attn(const bf16_t* __restrict__ Q, const bf16_t* __restrict__ K,
                const bf16_t* __restrict__ Vt, bf16_t* __restrict__ attn_out) {
    __shared__ bf16_t lds_p[4][32 * 32];   // per-wave 32x32 P tile (2KB each)

    const int lane = threadIdx.x & 31;
    const int wave = threadIdx.x >> 5;
    const int q0 = (blockIdx.x * 4 + wave) * 32;
    const int bh = blockIdx.y;

    const bf16_t* Qh = Q  + (size_t)bh * SEQ * D_K;
    const bf16_t* Kh = K  + (size_t)bh * SEQ * D_K;
    const bf16_t* Vh = Vt + (size_t)bh * D_K * SEQ;
    bf16_t* ldsP = lds_p[wave];
    const v16bf bones = ones_frag();

    v16bf aq[2][2];
#pragma unroll
    for (int mi = 0; mi < 2; ++mi) {
        aq[mi][0] = load_frag(Qh + (size_t)(q0 + mi * 16) * D_K +  0, D_K, lane);
        aq[mi][1] = load_frag(Qh + (size_t)(q0 + mi * 16) * D_K + 32, D_K, lane);
    }

    float rmax[2][8];
    v8f acc[2][4] = {};
    v8f accS[2] = {};
#pragma unroll
    for (int mi = 0; mi < 2; ++mi)
#pragma unroll
        for (int j = 0; j < 8; ++j) rmax[mi][j] = -__builtin_inff();

    // q0 is 32-aligned: blocks j0 < q0 are fully unmasked; j0 == q0 is the
    // only diagonal (masked) block.
    for (int j0 = 0; j0 < q0; j0 += 32)
        attn_block<false>(j0, q0, lane, Kh, Vh, aq, bones, ldsP, rmax, acc, accS);
    attn_block<true>(q0, q0, lane, Kh, Vh, aq, bones, ldsP, rmax, acc, accS);

    // normalize (v_rcp_f32 + mul) and store rows in [B*S, D_MODEL]
    const int bidx = bh >> 4, h = bh & (N_HEADS - 1);
    const int half = lane >> 4, n = lane & 15;
#pragma unroll
    for (int mi = 0; mi < 2; ++mi) {
        v8f inv;
#pragma unroll
        for (int j = 0; j < 8; ++j) inv[j] = __builtin_amdgcn_rcpf(accS[mi][j]);
#pragma unroll
        for (int c = 0; c < 4; ++c)
#pragma unroll
            for (int j = 0; j < 8; ++j) {
                const int m = mi * 16 + j + 8 * half;
                const size_t row = (size_t)bidx * SEQ + q0 + m;
                attn_out[row * D_MODEL + h * D_K + c * 16 + n] =
                    f2bf(acc[mi][c][j] * inv[j]);
            }
    }
}

// ---------------------------------------------------------------------------
// Kernel 3: output projection  O = attn @ W_o^T, fp32 out.
// ---------------------------------------------------------------------------
__global__ __launch_bounds__(128)
void out_gemm(const bf16_t* __restrict__ A, const bf16_t* __restrict__ W,
              float* __restrict__ out) {
    const int lane = threadIdx.x & 31;
    const int wave = threadIdx.x >> 5;
    const int m0 = blockIdx.y * 128 + wave * 32;
    const int n0 = blockIdx.x * 64;

    v8f acc[2][4] = {};
    gemm_tile_32x64(A + (size_t)m0 * D_MODEL, A + (size_t)(m0 + 16) * D_MODEL,
                    W + (size_t)n0 * D_MODEL, lane, acc);

    const int n = lane & 15, half = lane >> 4;
#pragma unroll
    for (int mi = 0; mi < 2; ++mi)
#pragma unroll
        for (int c = 0; c < 4; ++c)
#pragma unroll
            for (int j = 0; j < 8; ++j) {
                const int m = mi * 16 + j + 8 * half;
                out[(size_t)(m0 + m) * D_MODEL + n0 + c * 16 + n] = acc[mi][c][j];
            }
}

// ---------------------------------------------------------------------------
// Host-side launcher. Workspace layout (48 MiB total):
//   [0,8M)   x bf16            [8M,16M)  W_q/k/v/o bf16 (2M each)
//   [16,24M) Q(scaled) [bh][s][d]   [24,32M)  K [bh][s][d]
//   [32,40M) V^T [bh][d][s]    [40,48M)  attn [B*S][D_MODEL]
// ---------------------------------------------------------------------------
extern "C" void kernel_launch(void* const* d_in, const int* in_sizes, int n_in,
                              void* d_out, int out_size, void* d_ws, size_t ws_size,
                              hipStream_t stream) {
    (void)in_sizes; (void)n_in; (void)out_size; (void)ws_size;
    const float* x  = (const float*)d_in[0];
    const float* Wq = (const float*)d_in[1];
    const float* Wk = (const float*)d_in[2];
    const float* Wv = (const float*)d_in[3];
    const float* Wo = (const float*)d_in[4];

    char* ws = (char*)d_ws;
    const size_t MB = 1u << 20;
    bf16_t* xb   = (bf16_t*)(ws +  0 * MB);
    bf16_t* wqb  = (bf16_t*)(ws +  8 * MB);
    bf16_t* wkb  = (bf16_t*)(ws + 10 * MB);
    bf16_t* wvb  = (bf16_t*)(ws + 12 * MB);
    bf16_t* wob  = (bf16_t*)(ws + 14 * MB);
    bf16_t* qws  = (bf16_t*)(ws + 16 * MB);
    bf16_t* kws  = (bf16_t*)(ws + 24 * MB);
    bf16_t* vtws = (bf16_t*)(ws + 32 * MB);
    bf16_t* attn = (bf16_t*)(ws + 40 * MB);

    cvt_f32_bf16<<<512, 256, 0, stream>>>(x,  xb,  M_TOTAL * D_MODEL / 8);
    cvt_f32_bf16<<<128, 256, 0, stream>>>(Wq, wqb, D_MODEL * D_MODEL / 8);
    cvt_f32_bf16<<<128, 256, 0, stream>>>(Wk, wkb, D_MODEL * D_MODEL / 8);
    cvt_f32_bf16<<<128, 256, 0, stream>>>(Wv, wvb, D_MODEL * D_MODEL / 8);
    cvt_f32_bf16<<<128, 256, 0, stream>>>(Wo, wob, D_MODEL * D_MODEL / 8);

    dim3 pg(N_HEADS, M_TOTAL / 128);
    proj_gemm<false><<<pg, 128, 0, stream>>>(xb, wqb, qws,  SM_SCALE);
    proj_gemm<false><<<pg, 128, 0, stream>>>(xb, wkb, kws,  1.0f);
    proj_gemm<true ><<<pg, 128, 0, stream>>>(xb, wvb, vtws, 1.0f);

    dim3 fg(SEQ / 32 / 4, BATCH * N_HEADS);
    flash_attn<<<fg, 128, 0, stream>>>(qws, kws, vtws, attn);

    dim3 og(D_MODEL / 64, M_TOTAL / 128);
    out_gemm<<<og, 128, 0, stream>>>(attn, wob, (float*)d_out);
}
